// Finder_66657892434151
// MI455X (gfx1250) — 8-bit matrix operand order repaired and checked by reading, compiled, not run on hardware
//
#include <hip/hip_runtime.h>

typedef __attribute__((ext_vector_type(16))) int   v16i;
typedef __attribute__((ext_vector_type(8)))  float v8f;

#define B_     128
#define L_     4096
#define E_     128
#define V_     128
#define FP8_ONE 0x38u

#define TCHUNK 128
#define NCHUNK 33
#define SROWS  256
#define XSTRIDE 132
#define SPAD   160

#define WS_KEYS   0
#define WS_SEP    1024
#define WS_COL    1536
#define WS_IDS    2048
#define WS_FILT   67584

__global__ void k0_init(unsigned long long* keys, int* col) {
    int tid = threadIdx.x;
    if (tid < B_) keys[tid] = 0ull;
    if (tid == 0) col[0] = 0;
}

__global__ void k1_ids(const float* __restrict__ expr, const int* __restrict__ sep_tok,
                       int* __restrict__ ids, int* __restrict__ sep_pos,
                       int* __restrict__ col) {
    int b = blockIdx.x;
    int p = threadIdx.x;
    __shared__ int s_sep;
    if (p == 0) s_sep = E_;
    __syncthreads();
    const float* row = expr + ((size_t)b * E_ + p) * V_;
    int best = 0; float bv = row[0];
    for (int v = 1; v < V_; ++v) { float x = row[v]; if (x > bv) { bv = x; best = v; } }
    ids[b * E_ + p] = best;
    if (best == sep_tok[0]) atomicMin(&s_sep, p);
    __syncthreads();
    if (p == 0) {
        int sp = (s_sep == E_) ? 0 : s_sep;
        sep_pos[b] = sp;
        atomicMax(col, sp);
    }
}

__global__ void k2_filt(const int* __restrict__ ids, const int* __restrict__ sep_pos,
                        const int* __restrict__ col, const int* __restrict__ dot_tok,
                        unsigned char* __restrict__ filt) {
    int b = blockIdx.x;
    int tid = threadIdx.x;
    int sepb = sep_pos[b];
    int cs   = col[0];
    int dot  = dot_tok[0];
    unsigned* out = (unsigned*)(filt + (size_t)b * E_ * V_);
    for (int k = 0; k < 32; ++k) {
        int dw  = tid + 128 * k;
        int row = dw >> 5;
        int v0  = (dw & 31) << 2;
        int ide = (row >= sepb) ? dot : ids[b * E_ + row];
        unsigned u = 0;
        if (row < cs) {
            if (ide == v0 + 0) u |= FP8_ONE;
            if (ide == v0 + 1) u |= FP8_ONE << 8;
            if (ide == v0 + 2) u |= FP8_ONE << 16;
            if (ide == v0 + 3) u |= FP8_ONE << 24;
        }
        out[dw] = u;
    }
}

__global__ void __launch_bounds__(256)
k_scores(const float* __restrict__ X, const unsigned char* __restrict__ filt,
         unsigned long long* __restrict__ keys) {
    __shared__ __align__(16) unsigned char xs[SROWS * XSTRIDE];
    __shared__ float spad[SPAD];
    __shared__ unsigned long long key_sm[TCHUNK];

    const int tid   = threadIdx.x;
    const int lane  = tid & 31;
    const int w     = tid >> 5;
    const int chunk = blockIdx.x;
    const int b     = blockIdx.y;
    const int t0    = chunk * TCHUNK;

    if (tid < SPAD) spad[tid] = 0.0f;

    const float4* X4 = (const float4*)X;
    for (int k = 0; k < (SROWS * V_ / 4) / 256; ++k) {
        int e  = tid + 256 * k;
        int r  = e >> 5;
        int c4 = (e & 31) << 2;
        int s  = t0 + r;
        unsigned u = 0;
        if (s < L_) {
            float4 xv = X4[((size_t)b * L_ + s) * (V_ / 4) + (e & 31)];
            u = (xv.x != 0.f ? FP8_ONE : 0u)
              | ((xv.y != 0.f ? FP8_ONE : 0u) << 8)
              | ((xv.z != 0.f ? FP8_ONE : 0u) << 16)
              | ((xv.w != 0.f ? FP8_ONE : 0u) << 24);
        } else if (s < L_ + E_) {
            u = (c4 == 0) ? FP8_ONE : 0u;
        }
        *(unsigned*)&xs[r * XSTRIDE + c4] = u;
    }
    __syncthreads();

    const int bn   = lane & 15;
    const int hi16 = (lane >> 4) << 4;
    const unsigned char* frow = filt + ((size_t)b * E_ + (16 * w + bn)) * V_;
    v16i bf;
#pragma unroll
    for (int j = 0; j < 16; ++j) {
        int off = 32 * (j >> 2) + hi16 + 4 * (j & 3);
        bf[j] = *(const int*)(frow + off);
    }

    const int arow = lane & 15;
    const int hi8  = (lane >> 4) << 3;
    const int mib  = (lane >> 4) << 3;
    const int ni   = lane & 15;
    float* const dbase = &spad[mib - ni + 15];

    const unsigned char* xrow = &xs[(16 * w + arow) * XSTRIDE];
    auto loadA = [&](const unsigned char* p) {
        v16i a;
#pragma unroll
        for (int j = 0; j < 16; ++j) {
            int off = 32 * (j >> 2) + 4 * (j & 3) + 2 * hi8;
            a[j] = *(const int*)(p + off);
        }
        return a;
    };

    v16i a = loadA(xrow);
#pragma unroll
    for (int st = 0; st < 9; ++st) {
        v8f c = {};
        v8f d = __builtin_amdgcn_wmma_f32_16x16x128_fp8_fp8(a, bf, (short)0, c,
                                                            false, false);
        if (st < 8) {
            xrow += 16 * XSTRIDE;
            a = loadA(xrow);
        }
#pragma unroll
        for (int r = 0; r < 8; ++r)
            atomicAdd(dbase + 16 * st + r, d[r]);
    }
    __syncthreads();

    if (tid < TCHUNK) {
        int t = t0 + tid;
        unsigned long long key = 0ull;
        if (t <= L_) {
            unsigned long long sc = (unsigned long long)(int)(spad[tid + 15] + 0.5f);
            key = (sc << 32) | (unsigned long long)(0xFFFFFFFFu - (unsigned)t);
        }
        key_sm[tid] = key;
    }
    __syncthreads();
    if (tid == 0) {
        unsigned long long best = 0ull;
        for (int i = 0; i < TCHUNK; ++i) best = best > key_sm[i] ? best : key_sm[i];
        atomicMax(&keys[b], best);
    }
}

__global__ void k3_out(const unsigned long long* __restrict__ keys,
                       float* __restrict__ out) {
    int b = threadIdx.x;
    if (b < B_) {
        unsigned t = 0xFFFFFFFFu - (unsigned)(keys[b] & 0xFFFFFFFFull);
        out[b] = (float)t;
    }
}

extern "C" void kernel_launch(void* const* d_in, const int* in_sizes, int n_in,
                              void* d_out, int out_size, void* d_ws, size_t ws_size,
                              hipStream_t stream) {
    (void)in_sizes; (void)n_in; (void)out_size; (void)ws_size;
    const float* X    = (const float*)d_in[0];
    const float* expr = (const float*)d_in[1];
    const int* sep_t  = (const int*)d_in[2];
    const int* dot_t  = (const int*)d_in[4];

    char* ws = (char*)d_ws;
    unsigned long long* keys = (unsigned long long*)(ws + WS_KEYS);
    int*  sep_pos = (int*)(ws + WS_SEP);
    int*  col     = (int*)(ws + WS_COL);
    int*  ids     = (int*)(ws + WS_IDS);
    unsigned char* filt = (unsigned char*)(ws + WS_FILT);

    k0_init<<<1, 256, 0, stream>>>(keys, col);
    k1_ids<<<B_, 128, 0, stream>>>(expr, sep_t, ids, sep_pos, col);
    k2_filt<<<B_, 128, 0, stream>>>(ids, sep_pos, col, dot_t, filt);
    k_scores<<<dim3(NCHUNK, B_), 256, 0, stream>>>(X, filt, keys);
    k3_out<<<1, 128, 0, stream>>>(keys, (float*)d_out);
}
